// Transformer_20985210208746
// MI455X (gfx1250) — matrix operand order repaired and checked by reading, not compiled, not run on hardware
//
#include <hip/hip_runtime.h>
#include <cstdint>

// ---------------- model constants ----------------
#define D_MODEL 512
#define FF_DIM  2048
#define NLAYERS 6
#define BATCHN  4
#define SEQLEN  512
#define VOCAB_T 32000
#define NTOK    (BATCHN * SEQLEN)   // 2048 rows

typedef __attribute__((ext_vector_type(16))) __bf16 v16bf;
typedef __attribute__((ext_vector_type(8)))  float  v8f;

// ---------------- GEMM: C = A @ B (+bias) (+relu), bf16 WMMA, fp32 accum ----
// Block tile: (AF*64) x 128, K-step 32. 8 waves = 4 M-waves x 2 N-waves;
// each wave computes (AF*16) x 64 via AF A-fragments x 4 B-fragments.
#define BN 128
#define BK 32
#define BKP 40   // padded LDS row stride (bf16 elems): 80 bytes, 16B aligned

union FragU { uint4 q[2]; v16bf v; };

template <int AF>
__global__ __launch_bounds__(256)
void gemm_bf16_kernel(const float* __restrict__ A, long long sAb, int lda,
                      const float* __restrict__ B, long long sBb, int ldb, int bNK,
                      const float* __restrict__ bias,
                      float* __restrict__ C, long long sCb, int ldc,
                      int K, int relu)
{
  constexpr int BMT   = AF * 64;   // block M tile
  constexpr int APASS = BMT / 32;  // A staging passes (float4 per thread)
  __shared__ alignas(16) __bf16 As[BMT][BKP];
  __shared__ alignas(16) __bf16 Bs[BN][BKP];

  const int bz = blockIdx.z;
  A += (long long)bz * sAb;
  B += (long long)bz * sBb;
  C += (long long)bz * sCb;

  const int n0 = blockIdx.x * BN;
  const int m0 = blockIdx.y * BMT;

  const int t    = threadIdx.x;
  const int wave = t >> 5;
  const int lane = t & 31;
  const int wm   = wave & 3;      // M-wave index
  const int wn   = wave >> 2;     // N-wave index (64-col slice)
  const int half = lane >> 4;     // K-half selector per WMMA layout
  const int l16  = lane & 15;

  // staging coordinates
  const int c4  = (t & 7) * 4;    // A / B-NK column (float4)
  const int rr8 = t >> 3;         // 0..31 staging row
  const int n4  = (t & 31) * 4;   // B-KN column group
  const int kk8 = t >> 5;         // 0..7  B-KN k-row

  v8f acc[AF][4];
  #pragma unroll
  for (int a = 0; a < AF; ++a)
    #pragma unroll
    for (int i = 0; i < 4; ++i) acc[a][i] = (v8f)0.0f;

  float4 pa[APASS], pb[4];

  auto gload = [&](int k0) {
    #pragma unroll
    for (int p = 0; p < APASS; ++p)
      pa[p] = *(const float4*)(A + (long long)(m0 + rr8 + p * 32) * lda + k0 + c4);
    if (bNK) {  // B stored N x K row-major (K-matrix for Q*K^T)
      #pragma unroll
      for (int p = 0; p < 4; ++p)
        pb[p] = *(const float4*)(B + (long long)(n0 + rr8 + p * 32) * ldb + k0 + c4);
    } else {    // B stored K x N row-major (weights)
      #pragma unroll
      for (int p = 0; p < 4; ++p)
        pb[p] = *(const float4*)(B + (long long)(k0 + kk8 + p * 8) * ldb + n0 + n4);
    }
  };

  auto lstore = [&]() {
    #pragma unroll
    for (int p = 0; p < APASS; ++p) {
      const int row = rr8 + p * 32;
      As[row][c4 + 0] = (__bf16)pa[p].x;
      As[row][c4 + 1] = (__bf16)pa[p].y;
      As[row][c4 + 2] = (__bf16)pa[p].z;
      As[row][c4 + 3] = (__bf16)pa[p].w;
    }
    if (bNK) {
      #pragma unroll
      for (int p = 0; p < 4; ++p) {
        const int row = rr8 + p * 32;
        Bs[row][c4 + 0] = (__bf16)pb[p].x;
        Bs[row][c4 + 1] = (__bf16)pb[p].y;
        Bs[row][c4 + 2] = (__bf16)pb[p].z;
        Bs[row][c4 + 3] = (__bf16)pb[p].w;
      }
    } else {    // transpose while writing to LDS
      #pragma unroll
      for (int p = 0; p < 4; ++p) {
        const int kr = kk8 + p * 8;
        Bs[n4 + 0][kr] = (__bf16)pb[p].x;
        Bs[n4 + 1][kr] = (__bf16)pb[p].y;
        Bs[n4 + 2][kr] = (__bf16)pb[p].z;
        Bs[n4 + 3][kr] = (__bf16)pb[p].w;
      }
    }
  };

  gload(0);  // software-pipeline prologue
  for (int k0 = 0; k0 < K; k0 += BK) {
    lstore();
    __syncthreads();
    if (k0 + BK < K) gload(k0 + BK);   // overlap next global fetch with WMMA

    // ---- load ALL fragments first (distinct regs -> single dscnt wait),
    //      then issue the WMMAs back-to-back ----
    FragU fa[AF], fb[4];
    #pragma unroll
    for (int a = 0; a < AF; ++a) {
      const __bf16* ap = &As[wm * (16 * AF) + a * 16 + l16][half * 8];
      fa[a].q[0] = *(const uint4*)(ap);        // K = base .. base+7
      fa[a].q[1] = *(const uint4*)(ap + 16);   // K = base+16 .. base+23
    }
    #pragma unroll
    for (int ns = 0; ns < 4; ++ns) {
      const __bf16* bp = &Bs[wn * 64 + ns * 16 + l16][half * 8];
      fb[ns].q[0] = *(const uint4*)(bp);
      fb[ns].q[1] = *(const uint4*)(bp + 16);
    }
    #pragma unroll
    for (int a = 0; a < AF; ++a)
      #pragma unroll
      for (int ns = 0; ns < 4; ++ns)
        acc[a][ns] = __builtin_amdgcn_wmma_f32_16x16x32_bf16(
            false, fa[a].v, false, fb[ns].v, (short)0, acc[a][ns], false, false);
    __syncthreads();
  }

  // ---- epilogue: bias, relu, store fp32 ----
  const int cbase = n0 + wn * 64 + l16;
  #pragma unroll
  for (int a = 0; a < AF; ++a) {
    #pragma unroll
    for (int ns = 0; ns < 4; ++ns) {
      const int cc = cbase + ns * 16;
      const float bv = bias ? bias[cc] : 0.0f;
      #pragma unroll
      for (int r = 0; r < 8; ++r) {
        const int rm = m0 + wm * (16 * AF) + a * 16 + half * 8 + r;
        float v = acc[a][ns][r] + bv;
        if (relu) v = fmaxf(v, 0.0f);
        C[(long long)rm * ldc + cc] = v;
      }
    }
  }
}

// ---------------- block reduction helper (256 threads, wave32) -------------
__device__ __forceinline__ float blk_reduce(float v, float* red, int domax)
{
  #pragma unroll
  for (int o = 16; o; o >>= 1) {
    float w = __shfl_xor(v, o, 32);
    v = domax ? fmaxf(v, w) : (v + w);
  }
  const int wid = threadIdx.x >> 5;
  if ((threadIdx.x & 31) == 0) red[wid] = v;
  __syncthreads();
  float r = red[threadIdx.x & 7];
  #pragma unroll
  for (int o = 4; o; o >>= 1) {
    float w = __shfl_xor(r, o, 32);
    r = domax ? fmaxf(r, w) : (r + w);
  }
  __syncthreads();
  return r;
}

// ---------------- masked softmax over rows of length SEQLEN ----------------
// mode 0: none; 1: pad mask from seq; 2: pad mask + causal (k > q)
__global__ __launch_bounds__(256)
void softmax_kernel(float* __restrict__ S, const int* __restrict__ seq,
                    const int* __restrict__ padp, int mode, int Sq, int Sk)
{
  __shared__ float red[8];
  const int row = blockIdx.x;
  const int b = row / Sq, q = row - b * Sq;
  float* p = S + (long long)row * Sk;
  const int pad = (mode != 0) ? padp[0] : 0;
  const int t = threadIdx.x;
  float v[2];
  #pragma unroll
  for (int i = 0; i < 2; ++i) {
    const int k = t + i * 256;
    float x = p[k];
    bool msk = false;
    if (mode == 1)      msk = (seq[b * Sk + k] == pad);
    else if (mode == 2) msk = (seq[b * Sk + k] == pad) || (k > q);
    v[i] = msk ? -1e9f : x;
  }
  const float m = blk_reduce(fmaxf(v[0], v[1]), red, 1);
  const float e0 = __expf(v[0] - m), e1 = __expf(v[1] - m);
  const float s = blk_reduce(e0 + e1, red, 0);
  const float inv = 1.0f / s;
  p[t]       = e0 * inv;
  p[t + 256] = e1 * inv;
}

// ---------------- LayerNorm (optionally fused residual add) ----------------
__global__ __launch_bounds__(256)
void layernorm_kernel(const float* __restrict__ x, const float* __restrict__ res,
                      const float* __restrict__ w, const float* __restrict__ b,
                      float* __restrict__ out)
{
  __shared__ float red[8];
  const long long base = (long long)blockIdx.x * D_MODEL;
  const int t = threadIdx.x;
  float v0 = x[base + t]       + (res ? res[base + t]       : 0.0f);
  float v1 = x[base + t + 256] + (res ? res[base + t + 256] : 0.0f);
  const float mean = blk_reduce(v0 + v1, red, 0) * (1.0f / D_MODEL);
  const float d0 = v0 - mean, d1 = v1 - mean;
  const float var = blk_reduce(d0 * d0 + d1 * d1, red, 0) * (1.0f / D_MODEL);
  const float rs = rsqrtf(var + 1e-5f);
  out[base + t]       = d0 * rs * w[t]       + b[t];
  out[base + t + 256] = d1 * rs * w[t + 256] + b[t + 256];
}

// ---------------- embedding lookup + sinusoidal positional encoding --------
__global__ __launch_bounds__(256)
void embed_kernel(const int* __restrict__ seq, const float* __restrict__ emb,
                  float* __restrict__ out, int seqLen)
{
  const int tok = blockIdx.x;
  const int s = tok % seqLen;
  const long long id = (long long)seq[tok];
  const int t = threadIdx.x;
  #pragma unroll
  for (int i = 0; i < 2; ++i) {
    const int e = t + i * 256;
    const int i2 = e & ~1;
    const float div = __expf((float)i2 * (-9.210340371976184f / (float)D_MODEL));
    const float ang = (float)s * div;
    const float pe = (e & 1) ? __cosf(ang) : __sinf(ang);
    out[(long long)tok * D_MODEL + e] = emb[id * D_MODEL + e] + pe;
  }
}

// ---------------- host orchestration ---------------------------------------
extern "C" void kernel_launch(void* const* d_in, const int* in_sizes, int n_in,
                              void* d_out, int out_size, void* d_ws, size_t ws_size,
                              hipStream_t stream)
{
  (void)in_sizes; (void)out_size; (void)ws_size;
  if (n_in < 50) return;

  const int* src_seq = (const int*)d_in[0];
  const int* tgt_seq = (const int*)d_in[1];
  const int* src_pad = (const int*)d_in[2];
  const int* tgt_pad = (const int*)d_in[3];

  // params leaves in JAX sorted-dict order ('W' < 'b' in ASCII):
  // dec{attn{Wk,Wo,Wq,Wv,bk,bo,bq,bv}, cattn{...}, ff1_b,ff1_w,ff2_b,ff2_w,
  //     ln1_b,ln1_w,ln2_b,ln2_w,ln3_b,ln3_w}, emb_src, emb_tgt,
  // enc{attn{...}, ff1_b,ff1_w,ff2_b,ff2_w, ln1_b,ln1_w,ln2_b,ln2_w}, fc_b, fc_w
  enum { P = 4 };
  const int DA_WK=P+0,  DA_WO=P+1,  DA_WQ=P+2,  DA_WV=P+3,
            DA_BK=P+4,  DA_BO=P+5,  DA_BQ=P+6,  DA_BV=P+7;
  const int DC_WK=P+8,  DC_WO=P+9,  DC_WQ=P+10, DC_WV=P+11,
            DC_BK=P+12, DC_BO=P+13, DC_BQ=P+14, DC_BV=P+15;
  const int D_FF1B=P+16, D_FF1W=P+17, D_FF2B=P+18, D_FF2W=P+19;
  const int D_LN1B=P+20, D_LN1W=P+21, D_LN2B=P+22, D_LN2W=P+23,
            D_LN3B=P+24, D_LN3W=P+25;
  const int EMB_SRC=P+26, EMB_TGT=P+27;
  const int EA_WK=P+28, EA_WO=P+29, EA_WQ=P+30, EA_WV=P+31,
            EA_BK=P+32, EA_BO=P+33, EA_BQ=P+34, EA_BV=P+35;
  const int E_FF1B=P+36, E_FF1W=P+37, E_FF2B=P+38, E_FF2W=P+39;
  const int E_LN1B=P+40, E_LN1W=P+41, E_LN2B=P+42, E_LN2W=P+43;
  const int FC_B=P+44, FC_W=P+45;

  auto F = [&](int i) { return (const float*)d_in[i]; };

  // workspace layout (fp32)
  const long long NTD = (long long)NTOK * D_MODEL;      // 1,048,576
  float* ws  = (float*)d_ws;
  float* X   = ws;                 // encoder stream / enc_out
  float* Y   = X  + NTD;           // decoder stream
  float* Qb  = Y  + NTD;
  float* Kb  = Qb + NTD;
  float* Vb  = Kb + NTD;
  float* AT  = Vb + NTD;           // attn context (P @ V)
  float* Tb  = AT + NTD;           // post-LN temp
  float* Ob  = Tb + NTD;           // projection output temp
  float* SC  = Ob + NTD;           // scores: B*S*S
  float* FFb = SC + (long long)BATCHN * SEQLEN * SEQLEN;  // NTOK*FF

  auto gemm = [&](const float* Ap, long long sA, int lda,
                  const float* Bp, long long sB, int ldb, int bNK,
                  const float* bias, float* Cp, long long sC, int ldc,
                  int M, int N, int K, int batch, int relu) {
    if ((M % 128 == 0) && N >= 1024) {  // big tiles for the large GEMMs
      dim3 g(N / BN, M / 128, batch);
      gemm_bf16_kernel<2><<<g, dim3(256), 0, stream>>>(
          Ap, sA, lda, Bp, sB, ldb, bNK, bias, Cp, sC, ldc, K, relu);
    } else {                            // smaller tiles -> more blocks
      dim3 g(N / BN, M / 64, batch);
      gemm_bf16_kernel<1><<<g, dim3(256), 0, stream>>>(
          Ap, sA, lda, Bp, sB, ldb, bNK, bias, Cp, sC, ldc, K, relu);
    }
  };

  const long long SD = (long long)SEQLEN * D_MODEL;
  const long long SS = (long long)SEQLEN * SEQLEN;

  auto attention = [&](const float* xin, const float* kvin,
                       const float* Wq, const float* bq,
                       const float* Wk, const float* bk,
                       const float* Wv, const float* bv,
                       const float* Wo, const float* bo,
                       int maskMode, const int* mseq, const int* mpad,
                       float* outO) {
    gemm(xin, 0, D_MODEL, Wq, 0, D_MODEL, 0, bq, Qb, 0, D_MODEL,
         NTOK, D_MODEL, D_MODEL, 1, 0);
    gemm(kvin, 0, D_MODEL, Wk, 0, D_MODEL, 0, bk, Kb, 0, D_MODEL,
         NTOK, D_MODEL, D_MODEL, 1, 0);
    gemm(kvin, 0, D_MODEL, Wv, 0, D_MODEL, 0, bv, Vb, 0, D_MODEL,
         NTOK, D_MODEL, D_MODEL, 1, 0);
    // sim = Q @ K^T  (B is N x K row-major -> bNK=1), batched over B
    gemm(Qb, SD, D_MODEL, Kb, SD, D_MODEL, 1, nullptr, SC, SS, SEQLEN,
         SEQLEN, SEQLEN, D_MODEL, BATCHN, 0);
    softmax_kernel<<<BATCHN * SEQLEN, 256, 0, stream>>>(SC, mseq, mpad,
                                                        maskMode, SEQLEN, SEQLEN);
    // ctx = P @ V, batched
    gemm(SC, SS, SEQLEN, Vb, SD, D_MODEL, 0, nullptr, AT, SD, D_MODEL,
         SEQLEN, D_MODEL, SEQLEN, BATCHN, 0);
    gemm(AT, 0, D_MODEL, Wo, 0, D_MODEL, 0, bo, outO, 0, D_MODEL,
         NTOK, D_MODEL, D_MODEL, 1, 0);
  };

  // embeddings + positional encodings
  embed_kernel<<<NTOK, 256, 0, stream>>>(src_seq, F(EMB_SRC), X, SEQLEN);
  embed_kernel<<<NTOK, 256, 0, stream>>>(tgt_seq, F(EMB_TGT), Y, SEQLEN);

  // ----------------------- encoder stack -----------------------
  for (int l = 0; l < NLAYERS; ++l) {
    const long long oDD = (long long)l * D_MODEL * D_MODEL;
    const long long oD  = (long long)l * D_MODEL;
    const long long oDF = (long long)l * D_MODEL * FF_DIM;
    const long long oFD = (long long)l * FF_DIM * D_MODEL;
    const long long oF  = (long long)l * FF_DIM;

    attention(X, X,
              F(EA_WQ)+oDD, F(EA_BQ)+oD, F(EA_WK)+oDD, F(EA_BK)+oD,
              F(EA_WV)+oDD, F(EA_BV)+oD, F(EA_WO)+oDD, F(EA_BO)+oD,
              1, src_seq, src_pad, Ob);
    layernorm_kernel<<<NTOK, 256, 0, stream>>>(Ob, X, F(E_LN1W)+oD, F(E_LN1B)+oD, Tb);
    gemm(Tb, 0, D_MODEL, F(E_FF1W)+oDF, 0, FF_DIM, 0, F(E_FF1B)+oF,
         FFb, 0, FF_DIM, NTOK, FF_DIM, D_MODEL, 1, 1);
    gemm(FFb, 0, FF_DIM, F(E_FF2W)+oFD, 0, D_MODEL, 0, F(E_FF2B)+oD,
         Ob, 0, D_MODEL, NTOK, D_MODEL, FF_DIM, 1, 0);
    layernorm_kernel<<<NTOK, 256, 0, stream>>>(Ob, Tb, F(E_LN2W)+oD, F(E_LN2B)+oD, X);
  }

  // ----------------------- decoder stack -----------------------
  for (int l = 0; l < NLAYERS; ++l) {
    const long long oDD = (long long)l * D_MODEL * D_MODEL;
    const long long oD  = (long long)l * D_MODEL;
    const long long oDF = (long long)l * D_MODEL * FF_DIM;
    const long long oFD = (long long)l * FF_DIM * D_MODEL;
    const long long oF  = (long long)l * FF_DIM;

    // masked self-attention
    attention(Y, Y,
              F(DA_WQ)+oDD, F(DA_BQ)+oD, F(DA_WK)+oDD, F(DA_BK)+oD,
              F(DA_WV)+oDD, F(DA_BV)+oD, F(DA_WO)+oDD, F(DA_BO)+oD,
              2, tgt_seq, tgt_pad, Ob);
    layernorm_kernel<<<NTOK, 256, 0, stream>>>(Ob, Y, F(D_LN1W)+oD, F(D_LN1B)+oD, Tb);
    // cross-attention over encoder output (no mask)
    attention(Tb, X,
              F(DC_WQ)+oDD, F(DC_BQ)+oD, F(DC_WK)+oDD, F(DC_BK)+oD,
              F(DC_WV)+oDD, F(DC_BV)+oD, F(DC_WO)+oDD, F(DC_BO)+oD,
              0, nullptr, nullptr, Ob);
    layernorm_kernel<<<NTOK, 256, 0, stream>>>(Ob, Tb, F(D_LN2W)+oD, F(D_LN2B)+oD, Y);
    // FFN
    gemm(Y, 0, D_MODEL, F(D_FF1W)+oDF, 0, FF_DIM, 0, F(D_FF1B)+oF,
         FFb, 0, FF_DIM, NTOK, FF_DIM, D_MODEL, 1, 1);
    gemm(FFb, 0, FF_DIM, F(D_FF2W)+oFD, 0, D_MODEL, 0, F(D_FF2B)+oD,
         Ob, 0, D_MODEL, NTOK, D_MODEL, FF_DIM, 1, 0);
    layernorm_kernel<<<NTOK, 256, 0, stream>>>(Ob, Y, F(D_LN3W)+oD, F(D_LN3B)+oD, Y);
  }

  // ----------------------- final vocab projection -----------------------
  gemm(Y, 0, D_MODEL, F(FC_W), 0, VOCAB_T, 0, F(FC_B),
       (float*)d_out, 0, VOCAB_T, NTOK, VOCAB_T, D_MODEL, 1, 0);
}
